// DynamicWeightProjection_32384053412066
// MI455X (gfx1250) — matrix operand order repaired and checked by reading, not compiled, not run on hardware
//
#include <hip/hip_runtime.h>
#include <hip/hip_bf16.h>
#include <math.h>

// ---------------------------------------------------------------------------
// DynamicWeightProjection for MI455X (gfx1250), wave32 + v_wmma_f32_16x16x32_bf16
//
// Shapes: B=2,T=4096 -> R=8192 rows; D=4096; N1=512 (G*C*K); N2=128 (4M); C=4
// out = [ w1n (2,097,152) | w2 (2,097,152) | dd (1,048,576) ] f32
//
// k1: fused X*[dw1|dd_p] bf16 WMMA GEMM, double-buffered LDS A-staging
//     (1 barrier per 32-wide k-step), GELU->H / tanh->dd epilogue.
// k2: per-c H*qkw WMMA GEMM + RMS-norm(w1) epilogue via LDS.
// ---------------------------------------------------------------------------

typedef __attribute__((ext_vector_type(16))) __bf16 v16bf;
typedef __attribute__((ext_vector_type(8)))  __bf16 v8bf;
typedef __attribute__((ext_vector_type(8)))  float  v8f;

#define DDIM     4096
#define KSTEPS1  128      // 4096 / 32
#define LDSA     40       // padded A-row stride (bf16): 80B, 16B-aligned, conflict-free

// -------------------- prep: pre-swizzle weights into WMMA B-fragment layout ----
// B-fragment k order used here (same as A loads) for 16x16x32 bf16 WMMA (K x N = 32 x 16):
//   lane L: N = nt*16 + (L&15); K = ks*32 + (L<16 ? 0 : 16) + e, e=0..15 contiguous
// WtT  : [nt(40)][ks(128)][lane(32)][e(16)] bf16
// qkwT : [c(4)][nt(8)][ks(4)][lane(32)][e(16)] bf16
__global__ __launch_bounds__(256) void dwp_prep(const float* __restrict__ dw1,
                                                const float* __restrict__ ddp,
                                                const float* __restrict__ qkw,
                                                __bf16* __restrict__ WtT,
                                                __bf16* __restrict__ qkwT) {
    int tid = blockIdx.x * 256 + threadIdx.x;
    const int WCH = 40 * 128 * 32;               // chunks for WtT
    if (tid < WCH) {
        int lane = tid & 31;
        int ks   = (tid >> 5) & 127;
        int nt   = tid >> 12;                    // 0..39
        int n    = nt * 16 + (lane & 15);
        int kb   = ks * 32 + ((lane < 16) ? 0 : 16);
        __bf16* o = WtT + (size_t)tid * 16;
        #pragma unroll
        for (int e = 0; e < 16; ++e) {
            int k = kb + e;
            float v = (n < 512) ? dw1[(size_t)k * 512 + n]
                                : ddp[(size_t)k * 128 + (n - 512)];
            o[e] = (__bf16)v;
        }
    } else if (tid < WCH + 4096) {
        int t    = tid - WCH;
        int lane = t & 31;
        int ks   = (t >> 5) & 3;
        int nt   = (t >> 7) & 7;
        int c    = t >> 10;                      // 0..3
        int n    = nt * 16 + (lane & 15);
        int kb   = ks * 32 + ((lane < 16) ? 0 : 16);
        __bf16* o = qkwT + (size_t)t * 16;
        #pragma unroll
        for (int e = 0; e < 16; ++e) {
            int k = kb + e;
            o[e] = (__bf16)qkw[((size_t)c * 128 + k) * 128 + n];
        }
    }
}

// -------------------- k1: fused X*[dw1|dd_p] GEMM + GELU/tanh epilogue --------
// grid (64, 10); block 256 = 8 waves (4 row-groups x 2 col-groups)
// block tile: 128 rows x 64 cols; wave tile: 32x32 (2x2 WMMA); K-step 32
// Double-buffered LDS A tile: one barrier per k-step; X loads for step k+1
// in flight during step k's WMMAs.
__global__ __launch_bounds__(256) void dwp_gemm1(const float* __restrict__ X,
                                                 const __bf16* __restrict__ WtT,
                                                 __bf16* __restrict__ Hg,
                                                 float* __restrict__ dd_out) {
    __shared__ __bf16 As[2][128 * LDSA];

    const int tid  = threadIdx.x;
    const int lane = tid & 31;
    const int wid  = tid >> 5;
    const int wy   = wid & 1;          // col group (0/1)
    const int wx   = wid >> 1;         // row group (0..3)
    const int rowBlk = blockIdx.x * 128;
    const int nBlk   = blockIdx.y * 64;

    v8f acc[2][2] = {};

    // A-staging assignment: thread t handles row t/2, 16 f32 at col (t&1)*16
    const int lr = tid >> 1;
    const int lc = (tid & 1) * 16;
    const float* xrow = X + (size_t)(rowBlk + lr) * DDIM + lc;
    __bf16* const dst = (__bf16*)As + (size_t)lr * LDSA + lc;   // same offset in both buffers

    // A-fragment k order used here (16-bit A, 16x32, same as WtT): lanes 0-15 K 0..15; lanes 16-31 K 16..31
    const int kbA  = (lane < 16) ? 0 : 16;
    const int mA0  = wx * 32 + (lane & 15);
    const int ntg0 = blockIdx.y * 4 + wy * 2;    // global n-tile of this wave's col group

    // prologue: stage k-step 0 into buffer 0
    {
        const float4* src = (const float4*)(xrow);
        float4 f0 = src[0], f1 = src[1], f2 = src[2], f3 = src[3];
        v8bf lo, hi;
        lo[0]=(__bf16)f0.x; lo[1]=(__bf16)f0.y; lo[2]=(__bf16)f0.z; lo[3]=(__bf16)f0.w;
        lo[4]=(__bf16)f1.x; lo[5]=(__bf16)f1.y; lo[6]=(__bf16)f1.z; lo[7]=(__bf16)f1.w;
        hi[0]=(__bf16)f2.x; hi[1]=(__bf16)f2.y; hi[2]=(__bf16)f2.z; hi[3]=(__bf16)f2.w;
        hi[4]=(__bf16)f3.x; hi[5]=(__bf16)f3.y; hi[6]=(__bf16)f3.z; hi[7]=(__bf16)f3.w;
        *(v8bf*)(dst)     = lo;
        *(v8bf*)(dst + 8) = hi;
    }

    for (int ks = 0; ks < KSTEPS1; ++ks) {
        const int cur = ks & 1;
        __syncthreads();   // As[cur] ready; all prior reads of As[cur^1] retired

        // issue next k-step's X loads immediately (in flight during WMMAs)
        float4 f0, f1, f2, f3;
        const bool more = (ks + 1) < KSTEPS1;
        if (more) {
            const float4* src = (const float4*)(xrow + (size_t)(ks + 1) * 32);
            f0 = src[0]; f1 = src[1]; f2 = src[2]; f3 = src[3];
        }

        // fragments + WMMA from As[cur]
        const __bf16* abase = As[cur];
        v16bf afrag[2], bfrag[2];
        #pragma unroll
        for (int ms = 0; ms < 2; ++ms) {
            const __bf16* p = abase + (mA0 + ms * 16) * LDSA + kbA;
            *((v8bf*)&afrag[ms])     = *(const v8bf*)(p);
            *((v8bf*)&afrag[ms] + 1) = *(const v8bf*)(p + 8);
        }
        #pragma unroll
        for (int ns = 0; ns < 2; ++ns) {
            const __bf16* p = WtT + ((((size_t)(ntg0 + ns)) * 128 + ks) * 32 + lane) * 16;
            *((v8bf*)&bfrag[ns])     = *(const v8bf*)(p);
            *((v8bf*)&bfrag[ns] + 1) = *(const v8bf*)(p + 8);
        }
        #pragma unroll
        for (int ms = 0; ms < 2; ++ms)
            #pragma unroll
            for (int ns = 0; ns < 2; ++ns)
                acc[ms][ns] = __builtin_amdgcn_wmma_f32_16x16x32_bf16(
                    false, afrag[ms], false, bfrag[ns], (short)0, acc[ms][ns], false, false);

        // convert + store next k-step into the other buffer (no barrier needed:
        // everyone's reads of As[cur^1] happened before this iteration's barrier)
        if (more) {
            v8bf lo, hi;
            lo[0]=(__bf16)f0.x; lo[1]=(__bf16)f0.y; lo[2]=(__bf16)f0.z; lo[3]=(__bf16)f0.w;
            lo[4]=(__bf16)f1.x; lo[5]=(__bf16)f1.y; lo[6]=(__bf16)f1.z; lo[7]=(__bf16)f1.w;
            hi[0]=(__bf16)f2.x; hi[1]=(__bf16)f2.y; hi[2]=(__bf16)f2.z; hi[3]=(__bf16)f2.w;
            hi[4]=(__bf16)f3.x; hi[5]=(__bf16)f3.y; hi[6]=(__bf16)f3.z; hi[7]=(__bf16)f3.w;
            __bf16* d = dst + (cur ^ 1) * (128 * LDSA);
            *(v8bf*)(d)     = lo;
            *(v8bf*)(d + 8) = hi;
        }
    }

    // epilogue: cols <512 -> gelu -> Hg (bf16); cols >=512 -> tanh -> dd (f32)
    // uniform per block: blockIdx.y < 8 -> H path, else dd path
    const int colL   = lane & 15;
    const int rowOff = (lane >> 4) * 8;
    #pragma unroll
    for (int ms = 0; ms < 2; ++ms) {
        #pragma unroll
        for (int ns = 0; ns < 2; ++ns) {
            const int nGlob = nBlk + wy * 32 + ns * 16 + colL;
            #pragma unroll
            for (int v = 0; v < 8; ++v) {
                const int row = rowBlk + wx * 32 + ms * 16 + rowOff + v;
                float x = acc[ms][ns][v];
                if (nGlob < 512) {
                    float g = 0.5f * x * (1.0f + erff(x * 0.70710678118654752f));
                    Hg[(size_t)row * 512 + nGlob] = (__bf16)g;
                } else {
                    dd_out[(size_t)row * 128 + (nGlob - 512)] = tanhf(x);
                }
            }
        }
    }
}

// -------------------- k2: dw = gelu(H) * qkw_c ; split; RMS-norm w1 ----------
// grid 512 blocks x 128 threads (4 waves); block = 16-row tile, wave w -> c=w
__global__ __launch_bounds__(128) void dwp_gemm2(const __bf16* __restrict__ Hg,
                                                 const __bf16* __restrict__ qkwT,
                                                 float* __restrict__ w1n,
                                                 float* __restrict__ w2) {
    __shared__ float buf[4][16][132];   // per-wave 16x128 tile, padded

    const int tid  = threadIdx.x;
    const int lane = tid & 31;
    const int c    = tid >> 5;          // wave id == C index
    const int rt   = blockIdx.x;        // 16-row tile

    v8f acc[8] = {};

    const int kbA  = (lane < 16) ? 0 : 16;
    const int mrow = rt * 16 + (lane & 15);
    const __bf16* Arow = Hg + (size_t)mrow * 512 + c * 128;

    #pragma unroll
    for (int ks = 0; ks < 4; ++ks) {
        v16bf a;
        *((v8bf*)&a)     = *(const v8bf*)(Arow + ks * 32 + kbA);
        *((v8bf*)&a + 1) = *(const v8bf*)(Arow + ks * 32 + kbA + 8);
        #pragma unroll
        for (int nt = 0; nt < 8; ++nt) {
            v16bf b;
            const __bf16* p = qkwT + ((((size_t)c * 8 + nt) * 4 + ks) * 32 + lane) * 16;
            *((v8bf*)&b)     = *(const v8bf*)(p);
            *((v8bf*)&b + 1) = *(const v8bf*)(p + 8);
            acc[nt] = __builtin_amdgcn_wmma_f32_16x16x32_bf16(
                false, a, false, b, (short)0, acc[nt], false, false);
        }
    }

    // spill the 16x128 tile to LDS
    const int colL   = lane & 15;
    const int rowOff = (lane >> 4) * 8;
    #pragma unroll
    for (int nt = 0; nt < 8; ++nt)
        #pragma unroll
        for (int v = 0; v < 8; ++v)
            buf[c][rowOff + v][nt * 16 + colL] = acc[nt][v];
    __syncthreads();

    // lane L: row rr=L&15; handles i=(L>>4) of w1 and i2=2+(L>>4) of w2
    const int rr  = lane & 15;
    const int i1  = lane >> 4;        // w1 group index (0/1)
    const int i2  = 2 + (lane >> 4);  // w2 source group (2/3)
    const int row = rt * 16 + rr;

    {   // w1: RMS norm (no scale) over M=32
        float s = 0.0f;
        #pragma unroll
        for (int m = 0; m < 32; ++m) { float x = buf[c][rr][i1 * 32 + m]; s += x * x; }
        float scale = rsqrtf(s * (1.0f / 32.0f) + 1e-6f);
        float* o = w1n + (((size_t)row * 4 + c) * 2 + i1) * 32;
        #pragma unroll
        for (int m = 0; m < 32; ++m) o[m] = buf[c][rr][i1 * 32 + m] * scale;
    }
    {   // w2: raw copy
        float* o = w2 + (((size_t)row * 4 + c) * 2 + (i2 - 2)) * 32;
        #pragma unroll
        for (int m = 0; m < 32; ++m) o[m] = buf[c][rr][i2 * 32 + m];
    }
}

// ---------------------------------------------------------------------------
extern "C" void kernel_launch(void* const* d_in, const int* in_sizes, int n_in,
                              void* d_out, int out_size, void* d_ws, size_t ws_size,
                              hipStream_t stream) {
    const float* X    = (const float*)d_in[0];   // query_vec [2,4096,4096]
    // d_in[1] = key_vec (unused by reference)
    const float* dw1  = (const float*)d_in[2];   // [4096,1,4,128]
    const float* qkw  = (const float*)d_in[3];   // [1,4,128,4,32]
    const float* ddp  = (const float*)d_in[4];   // [4096,1,128]

    float* out  = (float*)d_out;
    float* w1n  = out;                 // 2,097,152
    float* w2   = out + 2097152;       // 2,097,152
    float* ddo  = out + 4194304;       // 1,048,576

    // workspace layout (bytes)
    char* ws = (char*)d_ws;
    __bf16* WtT   = (__bf16*)(ws);                       // 640*4096 bf16 = 5,242,880 B
    __bf16* qkwT  = (__bf16*)(ws + 5242880);             // 65,536 bf16  =   131,072 B
    __bf16* Hg    = (__bf16*)(ws + 5242880 + 131072);    // 8192*512 bf16 = 8,388,608 B

    dwp_prep<<<656, 256, 0, stream>>>(dw1, ddp, qkw, WtT, qkwT);
    dwp_gemm1<<<dim3(64, 10), 256, 0, stream>>>(X, WtT, Hg, ddo);
    dwp_gemm2<<<512, 128, 0, stream>>>(Hg, qkwT, w1n, w2);
}
